// RecentFocusedModel_11605001634017
// MI455X (gfx1250) — matrix operand order repaired and checked by reading, not compiled, not run on hardware
//
#include <hip/hip_runtime.h>
#include <hip/hip_fp16.h>
#include <hip/hip_bf16.h>

typedef __attribute__((ext_vector_type(16))) _Float16 v16h;
typedef __attribute__((ext_vector_type(8)))  _Float16 v8h;
typedef __attribute__((ext_vector_type(8)))  float    v8f;

union V16U { v16h v; v8h h[2]; _Float16 e[16]; };

#define Bsz 512
#define Ssz 256
#define Dsz 256
#define NLOC 50000
#define TEMPD 192
#define KIH 448          // D + TEMP
#define N4D 1024         // 4*D
#define NEGV (-1000000000.0f)

__device__ __forceinline__ float sigm(float x) { return 1.0f / (1.0f + __expf(-x)); }
__device__ __forceinline__ float gelu(float x) { return 0.5f * x * (1.0f + erff(x * 0.70710678118654752f)); }

// ---------------------------------------------------------------------------
// Pack a weight W (Ntot x Ktot, row-major, f32) into WMMA-B f16 lane order.
// B tile (32x16, K x N): lane L holds column n = nt*16 + (L&15),
// K = kt*32 + (L>>4)*16 + e  for e = 0..15 (contiguous per lane -> 32B loads).
// ---------------------------------------------------------------------------
__global__ void pack_b_f16(const float* __restrict__ W, _Float16* __restrict__ dst,
                           int Ntot, int Ktot) {
    long idx = (long)blockIdx.x * blockDim.x + threadIdx.x;
    long total = (long)Ntot * Ktot;
    if (idx >= total) return;
    int KT = Ktot >> 5;
    int e    = (int)(idx & 15);
    int lane = (int)((idx >> 4) & 31);
    long rest = idx >> 9;
    int kt = (int)(rest % KT);
    long nt = rest / KT;
    int n = (int)nt * 16 + (lane & 15);
    int k = kt * 32 + (lane >> 4) * 16 + e;
    dst[idx] = (_Float16)W[(size_t)n * Ktot + k];
}

// seq_lens[b] = sum(mask[b,:]) - 1
__global__ void seqlen_kernel(const unsigned char* __restrict__ mask, int* __restrict__ slen) {
    int b = blockIdx.x * blockDim.x + threadIdx.x;
    if (b >= Bsz) return;
    int s = 0;
    for (int t = 0; t < Ssz; ++t) s += (mask[b * Ssz + t] != 0) ? 1 : 0;
    slen[b] = s - 1;
}

// ---------------------------------------------------------------------------
// LSTM: one block = 16 batch rows, loops all 256 timesteps.
// block = 512 threads (16 waves); each wave owns 4 n-tiles (64 gate columns).
// ---------------------------------------------------------------------------
#define G_OFF   0
#define C_OFF   65536
#define X_OFF   (65536 + 16384)
#define H_OFF   (X_OFF + 14336)
#define BI_OFF  (H_OFF + 8192)
#define SL_OFF  (BI_OFF + 4096)
#define K1_SMEM (SL_OFF + 64)

__global__ void lstm_kernel(const int* __restrict__ locations, const int* __restrict__ users,
                            const int* __restrict__ weekdays,  const int* __restrict__ start_mins,
                            const float* __restrict__ loc_table, const float* __restrict__ user_table,
                            const float* __restrict__ hour_table, const float* __restrict__ wd_table,
                            const _Float16* __restrict__ wih_p, const _Float16* __restrict__ whh_p,
                            const float* __restrict__ b_ih, const float* __restrict__ b_hh,
                            const int* __restrict__ slen, float* __restrict__ ctx_out) {
    extern __shared__ char smem[];
    float*    gL  = (float*)(smem + G_OFF);     // [16][1024] gate preacts
    float*    cL  = (float*)(smem + C_OFF);     // [16][256]  cell state
    _Float16* xL  = (_Float16*)(smem + X_OFF);  // [16][448]  step input (f16)
    _Float16* hL  = (_Float16*)(smem + H_OFF);  // [16][256]  hidden (f16)
    float*    bL  = (float*)(smem + BI_OFF);    // [1024]     fused bias
    int*      sL  = (int*)(smem + SL_OFF);      // [16]       seq lens

    const int tid  = threadIdx.x;
    const int b0   = blockIdx.x * 16;
    const int lane = tid & 31;
    const int wave = tid >> 5;
    const int hsel = lane >> 4;       // 0 or 1
    const int mrow = lane & 15;
    const int koff = hsel * 16;

    for (int i = tid; i < N4D; i += blockDim.x) bL[i] = b_ih[i] + b_hh[i];
    for (int i = tid; i < 16 * Dsz; i += blockDim.x) { cL[i] = 0.0f; hL[i] = (_Float16)0.0f; }
    if (tid < 16) sL[tid] = slen[b0 + tid];
    __syncthreads();

    for (int t = 0; t < Ssz; ++t) {
        // --- gather step features into LDS (f32 -> f16) ---
        for (int i = tid; i < 16 * KIH; i += blockDim.x) {
            int row = i / KIH, col = i - row * KIH;
            int b = b0 + row;
            float v;
            if (col < Dsz) {
                int loc = locations[b * Ssz + t];
                v = loc_table[(size_t)loc * Dsz + col];
            } else if (col < Dsz + 64) {
                int u = users[b * Ssz + t];
                v = user_table[(size_t)u * 64 + (col - Dsz)];
            } else if (col < Dsz + 128) {
                int hh = start_mins[b * Ssz + t] / 60;
                hh = hh < 0 ? 0 : (hh > 24 ? 24 : hh);
                v = hour_table[hh * 64 + (col - Dsz - 64)];
            } else {
                int w = weekdays[b * Ssz + t];
                v = wd_table[w * 64 + (col - Dsz - 128)];
            }
            xL[i] = (_Float16)v;
        }
        __syncthreads();

        // --- gate GEMM: (16 x 704) x (704 x 1024), 4 n-tiles per wave ---
        for (int nl = 0; nl < 4; ++nl) {
            int ntg = wave * 4 + nl;                     // 0..63
            v8f acc = {};
            // x @ W_ih^T part (K = 448)
            for (int kt = 0; kt < KIH / 32; ++kt) {
                V16U A, Bv;
                const _Float16* pa = xL + mrow * KIH + kt * 32 + koff;
                A.h[0] = *(const v8h*)pa;
                A.h[1] = *(const v8h*)(pa + 8);
                const _Float16* pb = wih_p + ((size_t)(ntg * (KIH / 32) + kt) * 32 + lane) * 16;
                Bv.h[0] = *(const v8h*)pb;
                Bv.h[1] = *(const v8h*)(pb + 8);
                acc = __builtin_amdgcn_wmma_f32_16x16x32_f16(false, A.v, false, Bv.v,
                                                             (short)0, acc, false, false);
            }
            // h @ W_hh^T part (K = 256)
            for (int kt = 0; kt < Dsz / 32; ++kt) {
                V16U A, Bv;
                const _Float16* pa = hL + mrow * Dsz + kt * 32 + koff;
                A.h[0] = *(const v8h*)pa;
                A.h[1] = *(const v8h*)(pa + 8);
                const _Float16* pb = whh_p + ((size_t)(ntg * (Dsz / 32) + kt) * 32 + lane) * 16;
                Bv.h[0] = *(const v8h*)pb;
                Bv.h[1] = *(const v8h*)(pb + 8);
                acc = __builtin_amdgcn_wmma_f32_16x16x32_f16(false, A.v, false, Bv.v,
                                                             (short)0, acc, false, false);
            }
            int n = ntg * 16 + mrow;
            float bias = bL[n];
            #pragma unroll
            for (int r = 0; r < 8; ++r) {
                int m = r + 8 * hsel;
                gL[m * N4D + n] = acc[r] + bias;
            }
        }
        __syncthreads();

        // --- elementwise gates, state update ---
        for (int i = tid; i < 16 * Dsz; i += blockDim.x) {
            int m = i >> 8, d = i & 255;
            float gi = gL[m * N4D + d];
            float gf = gL[m * N4D + 256 + d];
            float gg = gL[m * N4D + 512 + d];
            float go = gL[m * N4D + 768 + d];
            float c  = sigm(gf) * cL[i] + sigm(gi) * tanhf(gg);
            float h  = sigm(go) * tanhf(c);
            cL[i] = c;
            hL[i] = (_Float16)h;
            if (t == sL[m]) ctx_out[(size_t)(b0 + m) * Dsz + d] = h;
        }
        __syncthreads();
    }
}

// ---------------------------------------------------------------------------
// Per-row head: mix gate, hidden = gelu(ctx @ np_W1^T + b1), recent scoring.
// One block (256 threads) per batch row.
// ---------------------------------------------------------------------------
__global__ void head_kernel(const float* __restrict__ context,
                            const float* __restrict__ np_W1, const float* __restrict__ np_b1,
                            const float* __restrict__ mx_W,  const float* __restrict__ mx_b,
                            const int* __restrict__ locations, const int* __restrict__ start_mins,
                            const float* __restrict__ loc_table,
                            const float* __restrict__ sc_W1, const float* __restrict__ sc_b1,
                            const float* __restrict__ sc_W2, const float* __restrict__ sc_b2,
                            const int* __restrict__ slen,
                            float* __restrict__ hid, _Float16* __restrict__ hidf16,
                            float* __restrict__ mix_out,
                            int* __restrict__ rl_loc, float* __restrict__ rl_score) {
    __shared__ float ctx_s[Dsz];
    __shared__ float red[256];
    __shared__ float a1[5 * 128];
    __shared__ int   s_rloc[5];
    __shared__ float s_tdiff[5];
    __shared__ int   s_valid[5];
    __shared__ float s_score[5];

    const int b = blockIdx.x;
    const int tid = threadIdx.x;

    ctx_s[tid] = context[(size_t)b * Dsz + tid];
    if (tid < 5) {
        int j = tid;
        int sl = slen[b];
        int idx = sl - j;
        int valid = idx >= 0;
        int idxc = idx < 0 ? 0 : (idx > Ssz - 1 ? Ssz - 1 : idx);
        s_rloc[j]  = locations[b * Ssz + idxc];
        int rtime  = start_mins[b * Ssz + idxc];
        int cur    = start_mins[b * Ssz + sl];
        s_tdiff[j] = (float)(cur - rtime) / 1440.0f;
        s_valid[j] = valid;
    }
    __syncthreads();

    // hidden layer of new-prediction head
    {
        int d = tid;
        float s = np_b1[d];
        const float* w = np_W1 + (size_t)d * Dsz;
        for (int k = 0; k < Dsz; ++k) s += ctx_s[k] * w[k];
        float g = gelu(s);
        hid[(size_t)b * Dsz + d] = g;
        hidf16[(size_t)b * Dsz + d] = (_Float16)g;
    }

    // mix = sigmoid(ctx . mx_W + mx_b) * 0.9 + 0.05
    red[tid] = ctx_s[tid] * mx_W[tid];
    __syncthreads();
    for (int off = 128; off > 0; off >>= 1) {
        if (tid < off) red[tid] += red[tid + off];
        __syncthreads();
    }
    if (tid == 0) mix_out[b] = sigm(red[0] + mx_b[0]) * 0.9f + 0.05f;
    __syncthreads();

    // recent scores: layer1 (258 -> 128), gelu
    for (int task = tid; task < 5 * 128; task += 256) {
        int j = task >> 7, u = task & 127;
        const float* w = sc_W1 + (size_t)u * 258;
        float s = sc_b1[u];
        const float* lt = loc_table + (size_t)s_rloc[j] * Dsz;
        for (int k = 0; k < Dsz; ++k) s += lt[k] * w[k];
        s += s_tdiff[j] * w[256] + (float)j * w[257];
        a1[j * 128 + u] = gelu(s);
    }
    __syncthreads();

    // layer2 (128 -> 1) per slot, 32 threads per slot
    float p = 0.0f;
    if (tid < 160) {
        int j = tid >> 5, l = tid & 31;
        for (int u = l; u < 128; u += 32) p += a1[j * 128 + u] * sc_W2[u];
    }
    red[tid] = p;
    __syncthreads();
    for (int off = 16; off > 0; off >>= 1) {
        if (tid < 160 && (tid & 31) < off) red[tid] += red[tid + off];
        __syncthreads();
    }
    if (tid < 5) s_score[tid] = red[tid * 32] + sc_b2[0];
    __syncthreads();

    // dedup (entry j dropped if a more-recent valid entry k<j has same loc)
    if (tid < 5) {
        int j = tid;
        int keep = s_valid[j];
        for (int k = 0; k < j; ++k)
            if (s_valid[k] && s_rloc[k] == s_rloc[j]) keep = 0;
        rl_loc[b * 5 + j]   = keep ? s_rloc[j] : -1;
        rl_score[b * 5 + j] = s_score[j];
    }
}

// ---------------------------------------------------------------------------
// out[b,n] = (1-mix[b]) * (hid[b] . np_W2[n] + np_b2[n]) + mix[b] * NEG
// WMMA f16 GEMM (512 x 50000, K=256). grid = (391 n-chunks, 32 b-tiles).
// ---------------------------------------------------------------------------
__global__ void outgemm_kernel(const _Float16* __restrict__ hidf16,
                               const _Float16* __restrict__ w2_p,
                               const float* __restrict__ np_b2,
                               const float* __restrict__ mix,
                               float* __restrict__ out) {
    __shared__ _Float16 hid_s[16 * Dsz];
    __shared__ float mix_s[16];

    const int tid  = threadIdx.x;
    const int b0   = blockIdx.y * 16;
    const int lane = tid & 31;
    const int wave = tid >> 5;
    const int hsel = lane >> 4;
    const int mrow = lane & 15;
    const int koff = hsel * 16;
    const int NT   = NLOC / 16;   // 3125

    for (int i = tid; i < 16 * Dsz; i += blockDim.x) {
        int row = i >> 8, col = i & 255;
        hid_s[i] = hidf16[(size_t)(b0 + row) * Dsz + col];
    }
    if (tid < 16) mix_s[tid] = mix[b0 + tid];
    __syncthreads();

    int ntg = blockIdx.x * 8 + wave;
    if (ntg >= NT) return;

    v8f acc = {};
    for (int kt = 0; kt < Dsz / 32; ++kt) {
        V16U A, Bv;
        const _Float16* pa = hid_s + mrow * Dsz + kt * 32 + koff;
        A.h[0] = *(const v8h*)pa;
        A.h[1] = *(const v8h*)(pa + 8);
        const _Float16* pb = w2_p + ((size_t)(ntg * (Dsz / 32) + kt) * 32 + lane) * 16;
        Bv.h[0] = *(const v8h*)pb;
        Bv.h[1] = *(const v8h*)(pb + 8);
        acc = __builtin_amdgcn_wmma_f32_16x16x32_f16(false, A.v, false, Bv.v,
                                                     (short)0, acc, false, false);
    }

    int n = ntg * 16 + mrow;
    float b2 = np_b2[n];
    #pragma unroll
    for (int r = 0; r < 8; ++r) {
        int m = r + 8 * hsel;
        float mx = mix_s[m];
        out[(size_t)(b0 + m) * NLOC + n] = (1.0f - mx) * (acc[r] + b2) + mx * NEGV;
    }
}

// ---------------------------------------------------------------------------
// Overwrite kept recent slots: out[b,loc] = mix*score + (1-mix)*new_logit.
// ---------------------------------------------------------------------------
__global__ void fixup_kernel(const int* __restrict__ rl_loc, const float* __restrict__ rl_score,
                             const float* __restrict__ hid, const float* __restrict__ np_W2,
                             const float* __restrict__ np_b2, const float* __restrict__ mix,
                             float* __restrict__ out) {
    __shared__ float red[256];
    const int b = blockIdx.x;
    const int tid = threadIdx.x;
    const int slot = tid >> 5;
    const int l = tid & 31;

    float p = 0.0f;
    int loc = -1;
    if (slot < 5) {
        loc = rl_loc[b * 5 + slot];
        if (loc >= 0) {
            const float* w = np_W2 + (size_t)loc * Dsz;
            const float* h = hid + (size_t)b * Dsz;
            for (int k = l; k < Dsz; k += 32) p += h[k] * w[k];
        }
    }
    red[tid] = p;
    __syncthreads();
    for (int off = 16; off > 0; off >>= 1) {
        if (l < off) red[tid] += red[tid + off];
        __syncthreads();
    }
    if (slot < 5 && l == 0 && loc >= 0) {
        float nl = red[slot * 32] + np_b2[loc];
        float mx = mix[b];
        out[(size_t)b * NLOC + loc] = mx * rl_score[b * 5 + slot] + (1.0f - mx) * nl;
    }
}

// ---------------------------------------------------------------------------
extern "C" void kernel_launch(void* const* d_in, const int* in_sizes, int n_in,
                              void* d_out, int out_size, void* d_ws, size_t ws_size,
                              hipStream_t stream) {
    const int*   locations  = (const int*)d_in[0];
    const int*   users      = (const int*)d_in[1];
    const int*   weekdays   = (const int*)d_in[2];
    const int*   start_mins = (const int*)d_in[3];
    const unsigned char* mask = (const unsigned char*)d_in[4];
    const float* loc_table  = (const float*)d_in[5];
    const float* user_table = (const float*)d_in[6];
    const float* hour_table = (const float*)d_in[7];
    const float* wd_table   = (const float*)d_in[8];
    const float* W_ih   = (const float*)d_in[9];
    const float* W_hh   = (const float*)d_in[10];
    const float* b_ih   = (const float*)d_in[11];
    const float* b_hh   = (const float*)d_in[12];
    const float* sc_W1  = (const float*)d_in[13];
    const float* sc_b1  = (const float*)d_in[14];
    const float* sc_W2  = (const float*)d_in[15];
    const float* sc_b2  = (const float*)d_in[16];
    const float* np_W1  = (const float*)d_in[17];
    const float* np_b1  = (const float*)d_in[18];
    const float* np_W2  = (const float*)d_in[19];
    const float* np_b2  = (const float*)d_in[20];
    const float* mx_W   = (const float*)d_in[21];
    const float* mx_b   = (const float*)d_in[22];
    float* out = (float*)d_out;

    // workspace layout (256-byte aligned slices)
    char* ws = (char*)d_ws;
    size_t off = 0;
    auto alloc = [&](size_t bytes) { char* p = ws + off; off += (bytes + 255) & ~(size_t)255; return p; };
    _Float16* wih_p  = (_Float16*)alloc((size_t)KIH * N4D * 2);     //  0.9 MB
    _Float16* whh_p  = (_Float16*)alloc((size_t)Dsz * N4D * 2);     //  0.5 MB
    _Float16* w2_p   = (_Float16*)alloc((size_t)Dsz * NLOC * 2);    // 25.6 MB
    float*    ctx    = (float*)alloc((size_t)Bsz * Dsz * 4);
    float*    hid    = (float*)alloc((size_t)Bsz * Dsz * 4);
    _Float16* hidf16 = (_Float16*)alloc((size_t)Bsz * Dsz * 2);
    int*      slen   = (int*)alloc(Bsz * 4);
    float*    mix    = (float*)alloc(Bsz * 4);
    int*      rl_loc = (int*)alloc(Bsz * 5 * 4);
    float*    rl_sc  = (float*)alloc(Bsz * 5 * 4);
    (void)ws_size; (void)in_sizes; (void)n_in; (void)out_size;

    // 1) pack weights to WMMA-B f16 lane order
    {
        long tot = (long)N4D * KIH;
        pack_b_f16<<<(tot + 255) / 256, 256, 0, stream>>>(W_ih, wih_p, N4D, KIH);
    }
    {
        long tot = (long)N4D * Dsz;
        pack_b_f16<<<(tot + 255) / 256, 256, 0, stream>>>(W_hh, whh_p, N4D, Dsz);
    }
    {
        long tot = (long)NLOC * Dsz;
        pack_b_f16<<<(tot + 255) / 256, 256, 0, stream>>>(np_W2, w2_p, NLOC, Dsz);
    }

    // 2) sequence lengths
    seqlen_kernel<<<2, 256, 0, stream>>>(mask, slen);

    // 3) LSTM scan (32 blocks x 512 threads, ~106 KB dynamic LDS each)
    lstm_kernel<<<Bsz / 16, 512, K1_SMEM, stream>>>(
        locations, users, weekdays, start_mins,
        loc_table, user_table, hour_table, wd_table,
        wih_p, whh_p, b_ih, b_hh, slen, ctx);

    // 4) per-row head: mix, hidden, recent scoring + dedup
    head_kernel<<<Bsz, 256, 0, stream>>>(
        ctx, np_W1, np_b1, mx_W, mx_b,
        locations, start_mins, loc_table,
        sc_W1, sc_b1, sc_W2, sc_b2, slen,
        hid, hidf16, mix, rl_loc, rl_sc);

    // 5) big output GEMM + mix combine -> d_out (covers every element)
    {
        dim3 grid((NLOC / 16 + 7) / 8, Bsz / 16);
        outgemm_kernel<<<grid, 256, 0, stream>>>(hidf16, w2_p, np_b2, mix, out);
    }

    // 6) recent-slot overwrite
    fixup_kernel<<<Bsz, 256, 0, stream>>>(rl_loc, rl_sc, hid, np_W2, np_b2, mix, out);
}
